// MultiheadAttention_4269197492394
// MI455X (gfx1250) — compile-verified, matrix operand order checked by reading, not run on hardware
//
#include <hip/hip_runtime.h>

#define BATCH 32
#define SEQ   2048
#define DK    128

typedef __attribute__((ext_vector_type(16))) _Float16 v16h;
typedef __attribute__((ext_vector_type(8)))  _Float16 v8h;
typedef __attribute__((ext_vector_type(8)))  float    v8f;

// 16-bit A/B fragment loader for V_WMMA_F32_16X16X32_F16 (wave32).
// Lane (h*16+m) holds row m of a row-major 16x32 tile:
//   elements 0..7  = K 8h..8h+7   (VGPRs 0-3)
//   elements 8..15 = K 16+8h..16+8h+7 (VGPRs 4-7)
// Both runs are 8 contiguous f16 => two 16-byte loads.
__device__ __forceinline__ v16h load_frag_rowmajor(const _Float16* base,
                                                   int ld, int m, int h) {
  const _Float16* p = base + (size_t)m * ld + 8 * h;
  v8h lo = *(const v8h*)(p);
  v8h hi = *(const v8h*)(p + 16);
  v16h r;
#pragma unroll
  for (int i = 0; i < 8; ++i) { r[i] = lo[i]; r[8 + i] = hi[i]; }
  return r;
}

// Convert K -> f16 row-major, V -> f16 transposed [B][D][S] so the P*V
// B-fragment (column of V) becomes a contiguous row load from V^T.
__global__ __launch_bounds__(256)
void cvt_kv_kernel(const float* __restrict__ K, const float* __restrict__ V,
                   _Float16* __restrict__ Kh, _Float16* __restrict__ Vt) {
  size_t i = (size_t)blockIdx.x * blockDim.x + threadIdx.x;
  if (i >= (size_t)BATCH * SEQ * DK) return;
  int d = (int)(i % DK);
  size_t bs = i / DK;
  int s = (int)(bs % SEQ);
  int b = (int)(bs / SEQ);
  Kh[i] = (_Float16)K[i];
  Vt[((size_t)b * DK + d) * SEQ + s] = (_Float16)V[i];
}

// One wave per (batch, 16-row query tile). 4 independent waves per block.
__global__ __launch_bounds__(128)
void attn_fused_kernel(const float* __restrict__ Q,
                       const _Float16* __restrict__ Kh,
                       const _Float16* __restrict__ Vt,
                       float* __restrict__ outR,
                       float* __restrict__ outA) {
  __shared__ _Float16 p_lds[4][16][40];  // padded stride 40 (80B, 16B-aligned)

  const int lane = threadIdx.x & 31;
  const int wave = threadIdx.x >> 5;
  const int m = lane & 15;
  const int h = lane >> 4;
  const int tile = blockIdx.x * 4 + wave;
  const int b  = tile >> 7;          // SEQ/16 == 128 tiles per batch
  const int qt = tile & 127;
  const int q0 = qt * 16;

  const _Float16* Kb = Kh + (size_t)b * SEQ * DK;
  const _Float16* Vb = Vt + (size_t)b * DK * SEQ;
  const float*    Qb = Q    + ((size_t)b * SEQ + q0) * DK;
  float*          Ab = outA + ((size_t)b * SEQ + q0) * SEQ;
  float*          Rb = outR + ((size_t)b * SEQ + q0) * DK;

  const float scale = 0.08838834764831845f;  // 1/sqrt(128)

  // Load + convert the Q tile (16x128) into 4 A-fragments (held in VGPRs).
  v16h qfrag[4];
#pragma unroll
  for (int kc = 0; kc < 4; ++kc) {
    const float* p = Qb + (size_t)m * DK + kc * 32 + 8 * h;
    v16h r;
#pragma unroll
    for (int i = 0; i < 8; ++i) {
      r[i]     = (_Float16)p[i];
      r[8 + i] = (_Float16)p[16 + i];
    }
    qfrag[kc] = r;
  }

  // ---- Pass A: softmax statistics via online max/sum (no score spill) ----
  // C layout: lane holds column n0+m; VGPR g is row g+8h.
  float rmax[8], rsum[8];
#pragma unroll
  for (int g = 0; g < 8; ++g) { rmax[g] = -3.0e38f; rsum[g] = 0.0f; }

  for (int nt = 0; nt < SEQ / 16; ++nt) {
    __builtin_prefetch(Kb + (size_t)(nt + 2) * 16 * DK, 0, 3);
    v8f c = {};
#pragma unroll
    for (int kc = 0; kc < 4; ++kc) {
      v16h bfrag = load_frag_rowmajor(Kb + (size_t)nt * 16 * DK + kc * 32, DK, m, h);
      c = __builtin_amdgcn_wmma_f32_16x16x32_f16(false, qfrag[kc], false, bfrag,
                                                 (short)0, c, false, false);
    }
#pragma unroll
    for (int g = 0; g < 8; ++g) {
      float x  = c[g] * scale;
      float nm = fmaxf(rmax[g], x);
      rsum[g]  = rsum[g] * __expf(rmax[g] - nm) + __expf(x - nm);
      rmax[g]  = nm;
    }
  }

  // Merge stats across the 16 lanes of each half-wave (same rows, diff cols).
#pragma unroll
  for (int mask = 1; mask <= 8; mask <<= 1) {
#pragma unroll
    for (int g = 0; g < 8; ++g) {
      float om = __shfl_xor(rmax[g], mask, 32);
      float os = __shfl_xor(rsum[g], mask, 32);
      float nm = fmaxf(rmax[g], om);
      rsum[g]  = rsum[g] * __expf(rmax[g] - nm) + os * __expf(om - nm);
      rmax[g]  = nm;
    }
  }
  float rinv[8];
#pragma unroll
  for (int g = 0; g < 8; ++g) rinv[g] = 1.0f / rsum[g];

  // ---- Pass B: recompute scores, normalize, emit attn once, accumulate P*V ----
  v8f zero = {};
  v8f oacc[8];
#pragma unroll
  for (int d0 = 0; d0 < 8; ++d0) oacc[d0] = zero;

  for (int nt = 0; nt < SEQ / 16; ++nt) {
    __builtin_prefetch(Kb + (size_t)(nt + 2) * 16 * DK, 0, 3);
    __builtin_prefetch(Vb + (size_t)nt * 16 + 2 * 32, 0, 3);
    v8f c = {};
#pragma unroll
    for (int kc = 0; kc < 4; ++kc) {
      v16h bfrag = load_frag_rowmajor(Kb + (size_t)nt * 16 * DK + kc * 32, DK, m, h);
      c = __builtin_amdgcn_wmma_f32_16x16x32_f16(false, qfrag[kc], false, bfrag,
                                                 (short)0, c, false, false);
    }
    const int col = nt * 16 + m;
#pragma unroll
    for (int g = 0; g < 8; ++g) {
      int   row = g + 8 * h;
      float p   = __expf(c[g] * scale - rmax[g]) * rinv[g];
      Ab[(size_t)row * SEQ + col] = p;                     // coalesced fp32 store
      p_lds[wave][row][(nt & 1) * 16 + m] = (_Float16)p;   // C-layout -> LDS
    }
    if (nt & 1) {
      // DS ops are in-order per wave, but make the store->load edge explicit.
      asm volatile("s_wait_dscnt 0" ::: "memory");
      v16h pa = load_frag_rowmajor(&p_lds[wave][0][0], 40, m, h);  // A-layout
      const int k0 = (nt - 1) * 16;
#pragma unroll
      for (int d0 = 0; d0 < 8; ++d0) {
        v16h vb = load_frag_rowmajor(Vb + (size_t)(d0 * 16) * SEQ + k0, SEQ, m, h);
        oacc[d0] = __builtin_amdgcn_wmma_f32_16x16x32_f16(false, pa, false, vb,
                                                          (short)0, oacc[d0],
                                                          false, false);
      }
    }
  }

  // Write the 16x128 result tile.
#pragma unroll
  for (int d0 = 0; d0 < 8; ++d0) {
#pragma unroll
    for (int g = 0; g < 8; ++g) {
      int row = g + 8 * h;
      Rb[(size_t)row * DK + d0 * 16 + m] = oacc[d0][g];
    }
  }
}

extern "C" void kernel_launch(void* const* d_in, const int* in_sizes, int n_in,
                              void* d_out, int out_size, void* d_ws, size_t ws_size,
                              hipStream_t stream) {
  // setup_inputs order: key, value, query (all fp32 [32, 2048, 128])
  const float* K = (const float*)d_in[0];
  const float* V = (const float*)d_in[1];
  const float* Q = (const float*)d_in[2];

  float* outR = (float*)d_out;                               // [B,S,D]
  float* outA = outR + (size_t)BATCH * SEQ * DK;             // [B,S,S]

  _Float16* Kh = (_Float16*)d_ws;                            // 16 MB
  _Float16* Vt = Kh + (size_t)BATCH * SEQ * DK;              // 16 MB (V^T)

  size_t n = (size_t)BATCH * SEQ * DK;
  cvt_kv_kernel<<<(unsigned)((n + 255) / 256), 256, 0, stream>>>(K, V, Kh, Vt);

  const int tiles = BATCH * (SEQ / 16);                      // 4096 waves
  attn_fused_kernel<<<tiles / 4, 128, 0, stream>>>(Q, Kh, Vt, outR, outA);
}
